// TransformerConv_3401614098972
// MI455X (gfx1250) — matrix operand order repaired and checked by reading, not compiled, not run on hardware
//
#include <hip/hip_runtime.h>
#include <hip/hip_bf16.h>

#define IN_F  128
#define OUT_F 128
#define HEADS 8
#define DH    16
#define COLS  512  // q(128) | k(128) | v(128) | self(128)

typedef __attribute__((ext_vector_type(16))) __bf16 v16bf;
typedef __attribute__((ext_vector_type(8)))  float  v8f;

union FragBF { v16bf v; unsigned int u[8]; };

__device__ __forceinline__ unsigned short f32_to_bf16(float f) {
  unsigned int u = __float_as_uint(f);
  u += 0x7fffu + ((u >> 16) & 1u);   // round-to-nearest-even
  return (unsigned short)(u >> 16);
}

__device__ __forceinline__ void atomicMaxF32(float* addr, float val) {
  int* ai = (int*)addr;
  int old = *ai;
  while (__int_as_float(old) < val) {
    int assumed = old;
    old = atomicCAS(ai, assumed, __float_as_int(val));
    if (old == assumed) break;
  }
}

// ---- pack weights into transposed bf16 [COLS][IN_F] + fused bias[COLS] ----
__global__ void k_pack_w(const float* __restrict__ Wq, const float* __restrict__ Wk,
                         const float* __restrict__ Wv, const float* __restrict__ Ws,
                         const float* __restrict__ bq, const float* __restrict__ bk,
                         const float* __restrict__ bv, const float* __restrict__ bs,
                         unsigned short* __restrict__ wbT, float* __restrict__ bias) {
  int idx = blockIdx.x * blockDim.x + threadIdx.x;
  if (idx < COLS) {
    int j = idx;
    float bval;
    if      (j < 128) bval = bq[j];
    else if (j < 256) bval = bk[j - 128];
    else if (j < 384) bval = bv[j - 256];
    else              bval = bs[j - 384];
    bias[j] = bval;
  }
  if (idx >= COLS * IN_F) return;
  int j = idx / IN_F, i = idx % IN_F;
  float w;
  if (j < 384) {
    int part = j >> 7;           // 0=q 1=k 2=v
    int jo = j & 127;
    int h = jo >> 4, d = jo & 15;
    const float* W = (part == 0) ? Wq : (part == 1) ? Wk : Wv;  // [H][IN][D]
    w = W[(h * IN_F + i) * DH + d];
  } else {
    w = Ws[i * OUT_F + (j - 384)];  // [IN][OUT]
  }
  wbT[j * IN_F + i] = f32_to_bf16(w);
}

__global__ void k_feat_bf16(const float* __restrict__ feat,
                            unsigned short* __restrict__ featb, int n) {
  int idx = blockIdx.x * blockDim.x + threadIdx.x;
  if (idx < n) featb[idx] = f32_to_bf16(feat[idx]);
}

__global__ void k_init(float* __restrict__ out, float* __restrict__ m,
                       float* __restrict__ den, int N) {
  int idx = blockIdx.x * blockDim.x + threadIdx.x;
  if (idx < N * OUT_F) out[idx] = 0.0f;
  if (idx < N * HEADS) { m[idx] = -3.402823466e38f; den[idx] = 0.0f; }
}

// ---- WMMA GEMM: QKVS[N][512] = bf16(feat[N][128]) @ WbT^T + bias ----
// One wave computes a 16-row tile across all 32 column tiles. The last row
// tile is clamped to start at N-16: overlapping rows are recomputed with
// identical values, so all loads and stores are unconditional (no exec-mask
// branching in the hot loop).
__global__ void __launch_bounds__(32)
k_gemm(const unsigned short* __restrict__ featb, const unsigned short* __restrict__ wbT,
       const float* __restrict__ bias, float* __restrict__ qkvs, int N) {
  const int lane = threadIdx.x & 31;
  const int l15  = lane & 15;
  const int hi   = (lane >> 4) & 1;
  int rowBase = blockIdx.x * 16;
  if (rowBase > N - 16) rowBase = N - 16;   // overlap tail tile (N >= 16)

  // ---- A fragments: 16 rows x K=128, 4 k-steps, held for the whole row tile
  const int r = rowBase + l15;
  const unsigned short* ap = featb + (size_t)r * IN_F;
  FragBF a[4];
  #pragma unroll
  for (int s = 0; s < 4; ++s) {
    // k order chosen here for A (contiguous, same as B): lanes0-15 -> K = kb..kb+15; lanes16-31 -> +16
    int kb = s * 32 + (hi ? 16 : 0);
    #pragma unroll
    for (int t = 0; t < 4; ++t) {
      a[s].u[t]     = *(const unsigned int*)(ap + kb + 2 * t);
      a[s].u[4 + t] = *(const unsigned int*)(ap + kb + 8 + 2 * t);
    }
  }

  for (int ct = 0; ct < 32; ++ct) {
    const int col = ct * 16 + l15;
    const unsigned short* bp = wbT + (size_t)col * IN_F;
    __builtin_prefetch(wbT + (size_t)(ct + 1) * 16 * IN_F, 0, 0);
    v8f acc = {0.f, 0.f, 0.f, 0.f, 0.f, 0.f, 0.f, 0.f};
    #pragma unroll
    for (int s = 0; s < 4; ++s) {
      // k order chosen here for B (contiguous, same as A): lanes0-15 (col=lane) hold K=kb..kb+15, lanes16-31 hold +16
      FragBF b;
      int kb = s * 32 + (hi ? 16 : 0);
      #pragma unroll
      for (int g = 0; g < 8; ++g)
        b.u[g] = *(const unsigned int*)(bp + kb + 2 * g);
      acc = __builtin_amdgcn_wmma_f32_16x16x32_bf16(
          false, a[s].v, false, b.v, (short)0, acc, false, false);
    }
    // C/D layout: VGPR t -> M = t (lanes0-15) / t+8 (lanes16-31), N = lane&15
    const float bj = bias[col];
    float* op = qkvs + (size_t)(rowBase + hi * 8) * COLS + col;
    #pragma unroll
    for (int t = 0; t < 8; ++t)
      op[(size_t)t * COLS] = acc[t] + bj;
  }
}

// ---- edge scores + segment max ----
__global__ void k_scores(const int* __restrict__ src, const int* __restrict__ dst,
                         const float* __restrict__ qkvs, float* __restrict__ scores,
                         float* __restrict__ m, int E) {
  int idx = blockIdx.x * blockDim.x + threadIdx.x;
  if (idx >= E * HEADS) return;
  int e = idx >> 3, h = idx & 7;
  int s = src[e], dn = dst[e];
  const float* kr = qkvs + (size_t)s  * COLS + 128 + h * DH;
  const float* qr = qkvs + (size_t)dn * COLS +       h * DH;
  float acc = 0.0f;
  #pragma unroll
  for (int t = 0; t < DH; ++t) acc += kr[t] * qr[t];
  acc *= 0.25f;  // 1/sqrt(16)
  scores[idx] = acc;
  atomicMaxF32(&m[dn * HEADS + h], acc);
}

// ---- exp(score - max) + segment sum ----
__global__ void k_exp(const int* __restrict__ dst, float* __restrict__ scores,
                      const float* __restrict__ m, float* __restrict__ den, int E) {
  int idx = blockIdx.x * blockDim.x + threadIdx.x;
  if (idx >= E * HEADS) return;
  int e = idx >> 3, h = idx & 7;
  int dn = dst[e];
  float ex = __expf(scores[idx] - m[dn * HEADS + h]);
  scores[idx] = ex;
  atomicAdd(&den[dn * HEADS + h], ex);
}

// ---- weighted scatter aggregation into d_out ----
__global__ void k_agg(const int* __restrict__ src, const int* __restrict__ dst,
                      const float* __restrict__ qkvs, const float* __restrict__ scores,
                      const float* __restrict__ den, float* __restrict__ out, int E) {
  int idx = blockIdx.x * blockDim.x + threadIdx.x;
  if (idx >= E * HEADS) return;
  int e = idx >> 3, h = idx & 7;
  int s = src[e], dn = dst[e];
  float w = scores[idx] / den[dn * HEADS + h];
  const float* vr = qkvs + (size_t)s * COLS + 256 + h * DH;
  float* op = out + (size_t)dn * OUT_F + h * DH;
  #pragma unroll
  for (int t = 0; t < DH; ++t) atomicAdd(&op[t], vr[t] * w);
}

// ---- add self projection residual ----
__global__ void k_final(const float* __restrict__ qkvs, float* __restrict__ out, int N) {
  int idx = blockIdx.x * blockDim.x + threadIdx.x;
  if (idx >= N * OUT_F) return;
  int n = idx >> 7, j = idx & 127;
  out[idx] += qkvs[(size_t)n * COLS + 384 + j];
}

extern "C" void kernel_launch(void* const* d_in, const int* in_sizes, int n_in,
                              void* d_out, int out_size, void* d_ws, size_t ws_size,
                              hipStream_t stream) {
  const float* feat = (const float*)d_in[0];
  const int*   src  = (const int*)d_in[1];
  const int*   dst  = (const int*)d_in[2];
  const float* Wq   = (const float*)d_in[3];
  const float* bq   = (const float*)d_in[4];
  const float* Wk   = (const float*)d_in[5];
  const float* bk   = (const float*)d_in[6];
  const float* Wv   = (const float*)d_in[7];
  const float* bv   = (const float*)d_in[8];
  const float* Ws   = (const float*)d_in[9];
  const float* bs   = (const float*)d_in[10];
  const int N = in_sizes[0] / IN_F;
  const int E = in_sizes[1];
  float* out = (float*)d_out;

  char* ws = (char*)d_ws;
  size_t off = 0;
  auto take = [&](size_t bytes) -> void* {
    void* p = ws + off;
    off += (bytes + 255) & ~(size_t)255;
    return p;
  };
  unsigned short* wbT   = (unsigned short*)take((size_t)COLS * IN_F * 2);
  float*          bias  = (float*)take((size_t)COLS * 4);
  unsigned short* featb = (unsigned short*)take((size_t)N * IN_F * 2);
  float*          qkvs  = (float*)take((size_t)N * COLS * 4);
  float*          mbuf  = (float*)take((size_t)N * HEADS * 4);
  float*          den   = (float*)take((size_t)N * HEADS * 4);
  float*          scr   = (float*)take((size_t)E * HEADS * 4);
  if (off > ws_size) return;  // insufficient workspace (~144 MB needed)

  const int EH = E * HEADS;
  k_pack_w   <<<(COLS * IN_F + 255) / 256, 256, 0, stream>>>(Wq, Wk, Wv, Ws, bq, bk, bv, bs, wbT, bias);
  k_feat_bf16<<<(N * IN_F + 255) / 256,    256, 0, stream>>>(feat, featb, N * IN_F);
  k_init     <<<(N * OUT_F + 255) / 256,   256, 0, stream>>>(out, mbuf, den, N);
  k_gemm     <<<(N + 15) / 16,              32, 0, stream>>>(featb, wbT, bias, qkvs, N);
  k_scores   <<<(EH + 255) / 256,          256, 0, stream>>>(src, dst, qkvs, scr, mbuf, E);
  k_exp      <<<(EH + 255) / 256,          256, 0, stream>>>(dst, scr, mbuf, den, E);
  k_agg      <<<(EH + 255) / 256,          256, 0, stream>>>(src, dst, qkvs, scr, den, out, E);
  k_final    <<<(N * OUT_F + 255) / 256,   256, 0, stream>>>(qkvs, out, N);
}
